// MultiScaleFusionBlock_64175401337032
// MI455X (gfx1250) — matrix operand order repaired and checked by reading, not compiled, not run on hardware
//
#include <hip/hip_runtime.h>

typedef __attribute__((ext_vector_type(16))) __bf16 bf16x16;
typedef __attribute__((ext_vector_type(8)))  __bf16 bf16x8;
typedef __attribute__((ext_vector_type(4)))  __bf16 bf16x4;
typedef __attribute__((ext_vector_type(8)))  float  f32x8;
typedef __attribute__((ext_vector_type(4)))  float  f32x4;

// global (address_space 1) pointer typedefs: force global_load/store instead of flat
typedef __attribute__((address_space(1))) const __bf16   gc_bf16;
typedef __attribute__((address_space(1))) const bf16x8   gc_bf16x8;
typedef __attribute__((address_space(1))) const bf16x16  gc_bf16x16;
typedef __attribute__((address_space(1))) const float    gc_f32;
typedef __attribute__((address_space(1))) const f32x4    gc_f32x4;
typedef __attribute__((address_space(1))) __bf16         g_bf16;
typedef __attribute__((address_space(1))) float          g_f32;

#define AS_G(T, p) ((T*)(unsigned long long)(p))

// ---------------- problem constants (from reference) ----------------
constexpr int BATCH = 32;
constexpr int CDIM  = 256;
constexpr int N1 = 64 * 64, N2 = 32 * 32, N3 = 16 * 16;
constexpr int NTOK = N1 + N2 + N3;          // 5376
constexpr int OFF1 = 0, OFF2 = N1, OFF3 = N1 + N2;

// =====================================================================
// LayerNorm (f32 in) -> bf16 out.  One wave (32 lanes) per token row.
// =====================================================================
__global__ __launch_bounds__(256)
void ln_to_bf16(const float* __restrict__ x, const float* __restrict__ g,
                const float* __restrict__ be, __bf16* __restrict__ out, int M)
{
    const int row  = blockIdx.x * 8 + (threadIdx.x >> 5);
    const int lane = threadIdx.x & 31;
    if (row >= M) return;
    gc_f32* xr = AS_G(gc_f32, x + (long)row * CDIM);
    float v[8], s = 0.f, s2 = 0.f;
#pragma unroll
    for (int i = 0; i < 8; ++i) {
        v[i] = xr[lane + 32 * i];
        s += v[i]; s2 += v[i] * v[i];
    }
#pragma unroll
    for (int off = 1; off < 32; off <<= 1) {
        s  += __shfl_xor(s,  off, 32);
        s2 += __shfl_xor(s2, off, 32);
    }
    const float mean = s * (1.f / 256.f);
    const float var  = s2 * (1.f / 256.f) - mean * mean;
    const float rstd = rsqrtf(var + 1e-5f);
    g_bf16* orow = AS_G(g_bf16, out + (long)row * CDIM);
#pragma unroll
    for (int i = 0; i < 8; ++i) {
        const int c = lane + 32 * i;
        orow[c] = (__bf16)((v[i] - mean) * rstd * g[c] + be[c]);
    }
}

// =====================================================================
// Bilinear 2x upsample (half-pixel centers, matches jax.image.resize)
// =====================================================================
__global__ __launch_bounds__(256)
void up2x(const __bf16* __restrict__ in, long inBatch, int IH, int IW,
          __bf16* __restrict__ out, int total)
{
    int idx = blockIdx.x * 256 + threadIdx.x;
    if (idx >= total) return;
    const int OW = IW * 2, OH = IH * 2;
    const int c = idx & 255;
    int r = idx >> 8;
    const int ox = r % OW; r /= OW;
    const int oy = r % OH; const int b = r / OH;
    const float sy = oy * 0.5f - 0.25f;
    const float sx = ox * 0.5f - 0.25f;
    int y0 = (int)floorf(sy); const float ty = sy - (float)y0;
    int x0 = (int)floorf(sx); const float tx = sx - (float)x0;
    int y1 = min(y0 + 1, IH - 1); y0 = max(y0, 0);
    int x1 = min(x0 + 1, IW - 1); x0 = max(x0, 0);
    gc_bf16* base = AS_G(gc_bf16, in) + (long)b * inBatch;
    const float v00 = (float)base[((long)y0 * IW + x0) * 256 + c];
    const float v01 = (float)base[((long)y0 * IW + x1) * 256 + c];
    const float v10 = (float)base[((long)y1 * IW + x0) * 256 + c];
    const float v11 = (float)base[((long)y1 * IW + x1) * 256 + c];
    const float v0 = v00 + (v01 - v00) * tx;
    const float v1 = v10 + (v11 - v10) * tx;
    AS_G(g_bf16, out)[((long)b * OH * OW + (long)oy * OW + ox) * 256 + c] =
        (__bf16)(v0 + (v1 - v0) * ty);
}

// 2x2 average downsample
__global__ __launch_bounds__(256)
void down2x(const __bf16* __restrict__ in, long inBatch, int IH, int IW,
            __bf16* __restrict__ out, int total)
{
    int idx = blockIdx.x * 256 + threadIdx.x;
    if (idx >= total) return;
    const int OW = IW / 2, OH = IH / 2;
    const int c = idx & 255;
    int r = idx >> 8;
    const int ox = r % OW; r /= OW;
    const int oy = r % OH; const int b = r / OH;
    gc_bf16* base = AS_G(gc_bf16, in) + (long)b * inBatch;
    float s = 0.f;
#pragma unroll
    for (int dy = 0; dy < 2; ++dy)
#pragma unroll
        for (int dx = 0; dx < 2; ++dx)
            s += (float)base[(((long)oy * 2 + dy) * IW + (ox * 2 + dx)) * 256 + c];
    AS_G(g_bf16, out)[((long)b * OH * OW + (long)oy * OW + ox) * 256 + c] =
        (__bf16)(s * 0.25f);
}

// =====================================================================
// WMMA GEMM:  out(b,p,n) = sum_k A(b,p,k) * W[n,k] + bias[n]
// A sourced from NCH (1 or 3) bf16 tensors, one per 256-wide K chunk
// (the multi-scale channel concat costs nothing).  LDS tiles live in
// WMMA *fragment* layout: staging = one 32B global load + two 16B ds
// stores per thread; each lane reads operands as aligned 32B vectors.
// Block = 256 threads = 8 waves; block tile 128(M) x 32(N); K step 32.
// Each wave keeps two accumulators (32x16) and reuses its B fragment.
// The K pipeline is FULLY UNROLLED (KTILES = NCH*8 is compile-time), so
// the per-chunk source-pointer select constant-folds -> no scratch.
// EPI 0: bf16 store.  EPI 1: f32 store with f32 residual add.
// =====================================================================
template<int EPI, int NCH>
__global__ __launch_bounds__(256)
void gemm_wmma(const __bf16* __restrict__ a0, long as0,
               const __bf16* __restrict__ a1, long as1,
               const __bf16* __restrict__ a2, long as2,
               int HW, int Ncols,
               const float* __restrict__ Wt,    // (Ncols, NCH*256) row-major f32
               const float* __restrict__ bias,  // (Ncols)
               __bf16* __restrict__ outB,
               const float* __restrict__ resF, long resBatch,
               float* __restrict__ outF, long outFBatch,
               int Mtot)
{
    constexpr int KTOT   = NCH * 256;
    constexpr int KTILES = NCH * 8;

    __shared__ __align__(32) __bf16 sA[2][8][32][16];   // 128x32 tile, frag layout
    __shared__ __align__(32) __bf16 sB[2][2][32][16];   // 32x32 tile, frag layout

    const int tid  = threadIdx.x;
    const int lane = tid & 31;
    const int wid  = tid >> 5;
    const int wm   = wid & 3;          // wave M pair 0..3 (covers 32 rows)
    const int wn   = wid >> 2;         // wave N tile 0..1
    const int lh   = lane >> 4;        // lane half
    const int ln   = lane & 15;
    const int nBase = blockIdx.x * 32;
    const int mBase = blockIdx.y * 128;

    f32x8 acc0 = {}, acc1 = {};

    // ---- A staging: thread covers 16 contiguous k of one row.
    // Fragment home of (row rt, k): lane = rt + 16*((k>>3)&1), j = (k&7)+8*(k>>4)
    // => halves [akk,akk+8) and [akk+8,akk+16) land at lanes rt / rt+16, same j0.
    const int arow = tid >> 1;              // 0..127
    const int akk  = (tid & 1) * 16;        // 0 or 16
    const int amt  = arow >> 4;             // fragment tile 0..7
    const int art  = arow & 15;
    const int aj0  = (akk >> 4) * 8;        // 0 or 8
    const int grow = mBase + arow;
    const bool rowok = grow < Mtot;
    int ab = 0, ap = 0;
    if (rowok) { ab = grow / HW; ap = grow - ab * HW; }

    // precomputed per-source row pointers (loop-invariant)
    gc_bf16* rp0 = AS_G(gc_bf16, a0) + (long)ab * as0 + (long)ap * 256 + akk;
    gc_bf16* rp1 = AS_G(gc_bf16, a1) + (long)ab * as1 + (long)ap * 256 + akk;
    gc_bf16* rp2 = AS_G(gc_bf16, a2) + (long)ab * as2 + (long)ap * 256 + akk;

    // ---- B staging: thread converts 4 contiguous f32 weights of one column.
    const int bn    = tid >> 3;             // 0..31
    const int bkk   = (tid & 7) * 4;        // 0..28
    const int bnt   = bn >> 4;
    const int blane = (bn & 15) + (((bkk >> 3) & 1) << 4);
    const int bj0   = (bkk & 7) + ((bkk >> 4) << 3);
    gc_f32* wp = AS_G(gc_f32, Wt) + (long)(nBase + bn) * KTOT + bkk;

    auto stage = [&](int kt) {              // kt is a compile-time constant here
        const int ch = kt >> 3;             // 256-wide source chunk (folds)
        gc_bf16* rp = (ch == 0) ? rp0 : ((ch == 1) ? rp1 : rp2);
        union { bf16x16 v; bf16x8 h[2]; } av;
        av.v = (bf16x16){};
        if (rowok) av.v = *(gc_bf16x16*)(rp + (kt & 7) * 32);
        *(bf16x8*)&sA[kt & 1][amt][art     ][aj0] = av.h[0];
        *(bf16x8*)&sA[kt & 1][amt][art + 16][aj0] = av.h[1];
        const f32x4 wv = *(gc_f32x4*)(wp + kt * 32);
        *(bf16x4*)&sB[kt & 1][bnt][blane][bj0] =
            (bf16x4){(__bf16)wv.x, (__bf16)wv.y, (__bf16)wv.z, (__bf16)wv.w};
    };

    stage(0);
#pragma unroll
    for (int kt = 0; kt < KTILES; ++kt) {
        __syncthreads();
        if (kt + 1 < KTILES) stage(kt + 1);
        const bf16x16 af0 = *(const bf16x16*)&sA[kt & 1][2 * wm    ][lane][0];
        const bf16x16 af1 = *(const bf16x16*)&sA[kt & 1][2 * wm + 1][lane][0];
        const bf16x16 bv  = *(const bf16x16*)&sB[kt & 1][wn][lane][0];
        acc0 = __builtin_amdgcn_wmma_f32_16x16x32_bf16(
                   false, af0, false, bv, (short)0, acc0, false, false);
        acc1 = __builtin_amdgcn_wmma_f32_16x16x32_bf16(
                   false, af1, false, bv, (short)0, acc1, false, false);
    }

#pragma unroll
    for (int half = 0; half < 2; ++half) {
        const f32x8 acc = half ? acc1 : acc0;
#pragma unroll
        for (int vr = 0; vr < 8; ++vr) {
            const int row = mBase + 32 * wm + 16 * half + vr + 8 * lh;  // C/D layout
            const int col = nBase + wn * 16 + ln;
            if (row >= Mtot) continue;
            const float v = acc[vr] + bias[col];
            if (EPI == 0) {
                AS_G(g_bf16, outB)[(long)row * Ncols + col] = (__bf16)v;
            } else {
                const int b = row / HW;
                const int p = row - b * HW;
                const long ri = (long)b * resBatch  + (long)p * 256 + col;
                const long oi = (long)b * outFBatch + (long)p * 256 + col;
                AS_G(g_f32, outF)[oi] = AS_G(gc_f32, resF)[ri] + v;
            }
        }
    }
}

// =====================================================================
// Windowed MHA: one block per 4x4 window, one wave per head.
// QK^T = one 16x16x32 WMMA; softmax across 16-lane halves (rows live in
// VGPR index + lane-half per C/D layout); P.V = two WMMAs (K padded).
// All fragments come from LDS as aligned 16/32B vector loads; V is
// pre-transposed once so the PV B fragment is a single 32B load.
// =====================================================================
template<int WDIM>
__global__ __launch_bounds__(256)
void wmha_wmma(const __bf16* __restrict__ qkv, const float* __restrict__ rpb,
               __bf16* __restrict__ ao)
{
    constexpr int HW  = WDIM * WDIM;
    constexpr int NWW = WDIM / 4;
    constexpr int NW  = NWW * NWW;
    __shared__ __align__(32) __bf16 sQKV[16][768];     // 24 KB
    __shared__ __align__(32) __bf16 sVt[8][32][16];    //  8 KB  V^T per head
    __shared__ __align__(32) __bf16 sPb[8][16][16];    //  4 KB  probs per head

    const int tid = threadIdx.x;
    const int b   = blockIdx.x / NW;
    const int wi  = blockIdx.x - b * NW;
    const int wy  = wi / NWW, wx = wi - wy * NWW;

    // stage the 16-token window's qkv rows into LDS (16B vector copies)
    for (int i = tid; i < 16 * 96; i += 256) {
        const int t  = i / 96;
        const int c8 = i - t * 96;
        const int gy = wy * 4 + (t >> 2);
        const int gx = wx * 4 + (t & 3);
        const long ro = ((long)b * HW + (long)gy * WDIM + gx) * 768;
        *(bf16x8*)&sQKV[t][c8 * 8] = AS_G(gc_bf16x8, qkv + ro)[c8];
    }
    __syncthreads();

    // transpose V into per-head (hd x nt) tiles
    for (int i = tid; i < 8 * 32 * 16; i += 256) {
        const int h = i >> 9, c = (i >> 4) & 31, t = i & 15;
        sVt[h][c][t] = sQKV[t][512 + h * 32 + c];
    }
    __syncthreads();

    const int h    = tid >> 5;         // head = wave
    const int lane = tid & 31;
    const int lh   = lane >> 4, ln = lane & 15;

    // Q fragment: A layout, 16 contiguous channels (k = 16*lh + j)
    union { bf16x16 v; bf16x8 q[2]; } qa;
    qa.q[0] = *(const bf16x8*)&sQKV[ln][h * 32 + 16 * lh];
    qa.q[1] = *(const bf16x8*)&sQKV[ln][h * 32 + 16 * lh + 8];
    // K^T fragment (B layout): 16 contiguous channels
    const bf16x16 kb = *(const bf16x16*)&sQKV[ln][256 + h * 32 + 16 * lh];

    f32x8 logit = {};
    logit = __builtin_amdgcn_wmma_f32_16x16x32_bf16(
                false, qa.v, false, kb, (short)0, logit, false, false);

    const float scale = 0.17677669529663687f;  // 32^-0.5
#pragma unroll
    for (int vr = 0; vr < 8; ++vr) {
        const int row = vr + 8 * lh;          // query token
        const int col = ln;                   // key token
        const int dy  = (row >> 2) - (col >> 2) + 3;
        const int dx  = (row & 3)  - (col & 3)  + 3;
        float l = logit[vr] * scale + rpb[(dy * 7 + dx) * 8 + h];
        float m = l;
#pragma unroll
        for (int off = 1; off < 16; off <<= 1) m = fmaxf(m, __shfl_xor(m, off, 16));
        const float e = __expf(l - m);
        float s = e;
#pragma unroll
        for (int off = 1; off < 16; off <<= 1) s += __shfl_xor(s, off, 16);
        sPb[h][row][col] = (__bf16)(e / s);   // own-head slice: intra-wave only
    }

    // P fragment: k=16*lh+j; lh=0 -> tokens 0..15 (valid), lh=1 -> k>=16 (zero pad)
    union { bf16x16 v; bf16x8 q[2]; } pa;
    pa.q[0] = lh ? (bf16x8){} : *(const bf16x8*)&sPb[h][ln][0];
    pa.q[1] = lh ? (bf16x8){} : *(const bf16x8*)&sPb[h][ln][8];

#pragma unroll
    for (int half = 0; half < 2; ++half) {
        bf16x16 vb = {};
        if (lh == 0)   // K rows 16..31 are the zero padding
            vb = *(const bf16x16*)&sVt[h][half * 16 + ln][0];
        f32x8 o = {};
        o = __builtin_amdgcn_wmma_f32_16x16x32_bf16(
                false, pa.v, false, vb, (short)0, o, false, false);
#pragma unroll
        for (int vr = 0; vr < 8; ++vr) {
            const int t  = vr + 8 * lh;
            const int gy = wy * 4 + (t >> 2);
            const int gx = wx * 4 + (t & 3);
            AS_G(g_bf16, ao)[((long)b * HW + (long)gy * WDIM + gx) * 256
                             + h * 32 + half * 16 + ln] = (__bf16)o[vr];
        }
    }
}

// =====================================================================
extern "C" void kernel_launch(void* const* d_in, const int* in_sizes, int n_in,
                              void* d_out, int out_size, void* d_ws, size_t ws_size,
                              hipStream_t stream) {
    const float* x       = (const float*)d_in[0];
    const float* n1g     = (const float*)d_in[1];
    const float* n1b     = (const float*)d_in[2];
    const float* conv_w[3] = {(const float*)d_in[3], (const float*)d_in[5], (const float*)d_in[7]};
    const float* conv_b[3] = {(const float*)d_in[4], (const float*)d_in[6], (const float*)d_in[8]};
    const float* qkv_w[3]  = {(const float*)d_in[9],  (const float*)d_in[14], (const float*)d_in[19]};
    const float* qkv_b[3]  = {(const float*)d_in[10], (const float*)d_in[15], (const float*)d_in[20]};
    const float* proj_w[3] = {(const float*)d_in[11], (const float*)d_in[16], (const float*)d_in[21]};
    const float* proj_b[3] = {(const float*)d_in[12], (const float*)d_in[17], (const float*)d_in[22]};
    const float* rpb[3]    = {(const float*)d_in[13], (const float*)d_in[18], (const float*)d_in[23]};
    const float* n3g     = (const float*)d_in[24];
    const float* n3b     = (const float*)d_in[25];
    const float* ffn_w   = (const float*)d_in[26];
    const float* ffn_b   = (const float*)d_in[27];
    float* yout = (float*)d_out;

    // ---------------- workspace carving ----------------
    char* w = (char*)d_ws;
    size_t off = 0;
    auto take = [&](size_t bytes) { void* p = w + off; off = (off + bytes + 255) & ~(size_t)255; return p; };
    const size_t TOKS = (size_t)BATCH * NTOK;
    __bf16* xn   = (__bf16*)take(TOKS * CDIM * 2);
    __bf16* a2u  = (__bf16*)take((size_t)BATCH * N1 * CDIM * 2);
    __bf16* t3u  = (__bf16*)take((size_t)BATCH * N2 * CDIM * 2);
    __bf16* t3uu = (__bf16*)take((size_t)BATCH * N1 * CDIM * 2);
    __bf16* dn1  = (__bf16*)take((size_t)BATCH * N2 * CDIM * 2);
    __bf16* dd1  = (__bf16*)take((size_t)BATCH * N3 * CDIM * 2);
    __bf16* dn2  = (__bf16*)take((size_t)BATCH * N3 * CDIM * 2);
    __bf16* fus[3] = {(__bf16*)take((size_t)BATCH * N1 * CDIM * 2),
                      (__bf16*)take((size_t)BATCH * N2 * CDIM * 2),
                      (__bf16*)take((size_t)BATCH * N3 * CDIM * 2)};
    __bf16* qkv[3] = {(__bf16*)take((size_t)BATCH * N1 * 768 * 2),
                      (__bf16*)take((size_t)BATCH * N2 * 768 * 2),
                      (__bf16*)take((size_t)BATCH * N3 * 768 * 2)};
    __bf16* ao[3]  = {(__bf16*)take((size_t)BATCH * N1 * CDIM * 2),
                      (__bf16*)take((size_t)BATCH * N2 * CDIM * 2),
                      (__bf16*)take((size_t)BATCH * N3 * CDIM * 2)};
    float*  ybuf = (float*)take(TOKS * CDIM * 4);
    __bf16* yn   = (__bf16*)take(TOKS * CDIM * 2);

    const long XNB = (long)NTOK * CDIM;   // batch stride inside xn / x / y

    // 1) LayerNorm -> bf16
    ln_to_bf16<<<(int)((TOKS + 7) / 8), 256, 0, stream>>>(x, n1g, n1b, xn, (int)TOKS);

    // 2) resampling pyramid
    {
        int t;
        t = BATCH * N1 * CDIM;
        up2x  <<<(t + 255) / 256, 256, 0, stream>>>(xn + (long)OFF2 * CDIM, XNB, 32, 32, a2u, t);
        t = BATCH * N2 * CDIM;
        up2x  <<<(t + 255) / 256, 256, 0, stream>>>(xn + (long)OFF3 * CDIM, XNB, 16, 16, t3u, t);
        t = BATCH * N1 * CDIM;
        up2x  <<<(t + 255) / 256, 256, 0, stream>>>(t3u, (long)N2 * CDIM, 32, 32, t3uu, t);
        t = BATCH * N2 * CDIM;
        down2x<<<(t + 255) / 256, 256, 0, stream>>>(xn, XNB, 64, 64, dn1, t);
        t = BATCH * N3 * CDIM;
        down2x<<<(t + 255) / 256, 256, 0, stream>>>(dn1, (long)N2 * CDIM, 32, 32, dd1, t);
        t = BATCH * N3 * CDIM;
        down2x<<<(t + 255) / 256, 256, 0, stream>>>(xn + (long)OFF2 * CDIM, XNB, 32, 32, dn2, t);
    }

    const int   HWs[3]  = {N1, N2, N3};
    const long  offs[3] = {OFF1, OFF2, OFF3};

    // 3) multi-scale fusion 1x1 convs (K = 3*256 via three A sources)
    {
        const __bf16* s1[3] = {xn + (long)OFF1 * CDIM, a2u, t3uu};
        const long    t1[3] = {XNB, (long)N1 * CDIM, (long)N1 * CDIM};
        const __bf16* s2[3] = {dn1, xn + (long)OFF2 * CDIM, t3u};
        const long    t2[3] = {(long)N2 * CDIM, XNB, (long)N2 * CDIM};
        const __bf16* s3[3] = {dd1, dn2, xn + (long)OFF3 * CDIM};
        const long    t3[3] = {(long)N3 * CDIM, (long)N3 * CDIM, XNB};
        const __bf16* const* S[3] = {s1, s2, s3};
        const long* T[3] = {t1, t2, t3};
        for (int s = 0; s < 3; ++s) {
            const int M = BATCH * HWs[s];
            dim3 grid(CDIM / 32, (M + 127) / 128);
            gemm_wmma<0, 3><<<grid, 256, 0, stream>>>(
                S[s][0], T[s][0], S[s][1], T[s][1], S[s][2], T[s][2],
                HWs[s], CDIM, conv_w[s], conv_b[s],
                fus[s], nullptr, 0, nullptr, 0, M);
        }
    }

    // 4) qkv projections (N = 768)
    for (int s = 0; s < 3; ++s) {
        const int M = BATCH * HWs[s];
        const long st = (long)HWs[s] * CDIM;
        dim3 grid(768 / 32, (M + 127) / 128);
        gemm_wmma<0, 1><<<grid, 256, 0, stream>>>(
            fus[s], st, fus[s], st, fus[s], st,
            HWs[s], 768, qkv_w[s], qkv_b[s],
            qkv[s], nullptr, 0, nullptr, 0, M);
    }

    // 5) windowed attention
    wmha_wmma<64><<<BATCH * (16 * 16), 256, 0, stream>>>(qkv[0], rpb[0], ao[0]);
    wmha_wmma<32><<<BATCH * (8 * 8),   256, 0, stream>>>(qkv[1], rpb[1], ao[1]);
    wmha_wmma<16><<<BATCH * (4 * 4),   256, 0, stream>>>(qkv[2], rpb[2], ao[2]);

    // 6) output proj + shortcut residual  ->  y (f32)
    for (int s = 0; s < 3; ++s) {
        const int M = BATCH * HWs[s];
        const long st = (long)HWs[s] * CDIM;
        dim3 grid(CDIM / 32, (M + 127) / 128);
        gemm_wmma<1, 1><<<grid, 256, 0, stream>>>(
            ao[s], st, ao[s], st, ao[s], st,
            HWs[s], CDIM, proj_w[s], proj_b[s],
            nullptr, x + offs[s] * CDIM, XNB,
            ybuf + offs[s] * CDIM, XNB, M);
    }

    // 7) LN(y) -> bf16
    ln_to_bf16<<<(int)((TOKS + 7) / 8), 256, 0, stream>>>(ybuf, n3g, n3b, yn, (int)TOKS);

    // 8) FFN + residual -> d_out
    {
        const int M = (int)TOKS;
        dim3 grid(CDIM / 32, (M + 127) / 128);
        gemm_wmma<1, 1><<<grid, 256, 0, stream>>>(
            yn, (long)NTOK * CDIM, yn, (long)NTOK * CDIM, yn, (long)NTOK * CDIM,
            NTOK, CDIM, ffn_w, ffn_b,
            nullptr, ybuf, XNB, yout, XNB, M);
    }
    (void)in_sizes; (void)n_in; (void)out_size; (void)ws_size;
}
